// CausalAttention_40312563040663
// MI455X (gfx1250) — compile-verified, matrix operand order checked by reading, not run on hardware
//
#include <hip/hip_runtime.h>
#include <hip/hip_bf16.h>

// ---------------------------------------------------------------------------
// CausalAttention forward for gfx1250 (MI455X): bf16 WMMA everywhere,
// flash-attention inner loop, async global->LDS tile staging (ASYNCcnt path).
//   B=4, T=2048, C=1024, H=16, hd=64, M = B*T = 8192
// ---------------------------------------------------------------------------

typedef __attribute__((ext_vector_type(16))) __bf16 bf16x16;
typedef __attribute__((ext_vector_type(8)))  __bf16 bf16x8;
typedef __attribute__((ext_vector_type(8)))  float  floatx8;
typedef __attribute__((ext_vector_type(8)))  unsigned short ushort8;

union Frag16 { bf16x16 v; bf16x8 h[2]; };

__device__ __forceinline__ unsigned short f2bf(float f) {
    unsigned int u = __builtin_bit_cast(unsigned int, f);
    u += 0x7FFFu + ((u >> 16) & 1u);          // round-to-nearest-even
    return (unsigned short)(u >> 16);
}

__device__ __forceinline__ floatx8 wmma_bf16(bf16x16 a, bf16x16 b, floatx8 c) {
    // v_wmma_f32_16x16x32_bf16
    return __builtin_amdgcn_wmma_f32_16x16x32_bf16(
        /*neg_a=*/false, a, /*neg_b=*/false, b,
        /*c_mod=*/(short)0, c, /*reuse_a=*/false, /*reuse_b=*/false);
}

// 16x32 bf16 A/B fragment load (ISA 16-bit A-matrix layout):
// lane L holds row (row0 + L%16); elements 0..7 -> k = 8*(L/16)+0..7,
// elements 8..15 -> k = 16 + 8*(L/16) + 0..7.  Two 16B loads.
__device__ __forceinline__ bf16x16 ldFrag(const unsigned short* base, int row0,
                                          int lane, int stride) {
    const unsigned short* p = base + (size_t)(row0 + (lane & 15)) * stride
                            + ((lane >> 4) << 3);
    Frag16 f;
    f.h[0] = *(const bf16x8*)(p);
    f.h[1] = *(const bf16x8*)(p + 16);
    return f.v;
}

// --- CDNA5 async global->LDS (tracked by ASYNCcnt, bypasses VGPRs) ---------
// LDS offset of a __shared__ object = low 32 bits of its flat address
// (ISA 10.2, LDS aperture).
__device__ __forceinline__ unsigned ldsOff(const void* p) {
    return (unsigned)(unsigned long long)p;
}
__device__ __forceinline__ void asyncLoadB128(unsigned lds_byte_off,
                                              const void* gaddr) {
    asm volatile("global_load_async_to_lds_b128 %0, %1, off"
                 :: "v"(lds_byte_off), "v"(gaddr) : "memory");
}
__device__ __forceinline__ void waitAsync() {
    asm volatile("s_wait_asynccnt 0" ::: "memory");
}

// ---------------------------------------------------------------------------
// fp32 -> bf16 bulk convert (n multiple of 1024)
// ---------------------------------------------------------------------------
__global__ __launch_bounds__(256) void cvt_f32_bf16(const float* __restrict__ src,
                                                    unsigned short* __restrict__ dst,
                                                    int n) {
    int i = (blockIdx.x * 256 + threadIdx.x) * 4;
    if (i + 3 < n) {
        float4 v = *(const float4*)(src + i);
        dst[i + 0] = f2bf(v.x);
        dst[i + 1] = f2bf(v.y);
        dst[i + 2] = f2bf(v.z);
        dst[i + 3] = f2bf(v.w);
    }
}

// src [K][N] fp32 (row-major)  ->  dst [N][K] bf16 (row-major, i.e. W^T)
__global__ __launch_bounds__(256) void transpose_cvt(const float* __restrict__ src,
                                                     unsigned short* __restrict__ dst,
                                                     int K, int N) {
    int i = blockIdx.x * 256 + threadIdx.x;
    if (i < N * K) {
        int n = i / K;
        int k = i - n * K;
        dst[i] = f2bf(src[(size_t)k * N + n]);
    }
}

// ---------------------------------------------------------------------------
// bf16 GEMM: C[M,N] = A[M,K] * Bt[N,K]^T, 128x128 block tile, BK=32,
// 8 waves (4x2), wave tile 32x64 (2x4 WMMA tiles, fp32 acc).
// Double-buffered LDS filled by global_load_async_to_lds_b128; the fetch of
// tile k+1 overlaps the 8 WMMAs of tile k.
// MODE 0: scatter into q/k/v buffers   MODE 1: write fp32 to outF [M,1024]
// ---------------------------------------------------------------------------
#define BM 128
#define BN 128
#define BK 32
#define LDT 40   // padded LDS row stride (shorts) -> conflict-free 16B reads

template <int MODE>
__global__ __launch_bounds__(256) void gemm_bf16(
    const unsigned short* __restrict__ A,    // [M x K] bf16
    const unsigned short* __restrict__ Bt,   // [N x K] bf16 (B transposed)
    float* __restrict__ outF,
    unsigned short* __restrict__ qb,
    unsigned short* __restrict__ kbuf,
    unsigned short* __restrict__ vb,
    int K) {
    __shared__ unsigned short As[2][BM * LDT];
    __shared__ unsigned short Bs[2][BN * LDT];

    const int tid  = threadIdx.x;
    const int lane = tid & 31;
    const int wave = tid >> 5;
    const int wm   = wave >> 1;              // 0..3
    const int wn   = wave & 1;               // 0..1
    const int m0   = blockIdx.x * BM;
    const int n0   = blockIdx.y * BN;

    floatx8 acc[2][4];
#pragma unroll
    for (int i = 0; i < 2; ++i)
#pragma unroll
        for (int j = 0; j < 4; ++j) acc[i][j] = (floatx8)0.0f;

    const int lrow = tid >> 1;               // 0..127
    const int lcol = (tid & 1) * 16;         // 0 or 16
    const unsigned asB = ldsOff(&As[0][0]) + (unsigned)(lrow * LDT + lcol) * 2u;
    const unsigned bsB = ldsOff(&Bs[0][0]) + (unsigned)(lrow * LDT + lcol) * 2u;
    const unsigned bufBytes = (unsigned)(BM * LDT) * 2u;
    const unsigned short* gaRow = A  + (size_t)(m0 + lrow) * K + lcol;
    const unsigned short* gbRow = Bt + (size_t)(n0 + lrow) * K + lcol;

    const int nk = K / BK;
    // prologue: stage tile 0 into buffer 0
    asyncLoadB128(asB,      gaRow);
    asyncLoadB128(asB + 16, gaRow + 8);
    asyncLoadB128(bsB,      gbRow);
    asyncLoadB128(bsB + 16, gbRow + 8);

    for (int it = 0; it < nk; ++it) {
        waitAsync();                 // my async copies for buf[it&1] done
        __syncthreads();             // tile visible; buf[(it+1)&1] free
        if (it + 1 < nk) {
            const unsigned nb = (unsigned)((it + 1) & 1) * bufBytes;
            const int k1 = (it + 1) * BK;
            asyncLoadB128(asB + nb,      gaRow + k1);
            asyncLoadB128(asB + nb + 16, gaRow + k1 + 8);
            asyncLoadB128(bsB + nb,      gbRow + k1);
            asyncLoadB128(bsB + nb + 16, gbRow + k1 + 8);
        }
        const unsigned short* as = As[it & 1];
        const unsigned short* bs = Bs[it & 1];
        bf16x16 af[2], bfv[4];
#pragma unroll
        for (int i = 0; i < 2; ++i) af[i]  = ldFrag(as, wm * 32 + i * 16, lane, LDT);
#pragma unroll
        for (int j = 0; j < 4; ++j) bfv[j] = ldFrag(bs, wn * 64 + j * 16, lane, LDT);
#pragma unroll
        for (int i = 0; i < 2; ++i)
#pragma unroll
            for (int j = 0; j < 4; ++j) acc[i][j] = wmma_bf16(af[i], bfv[j], acc[i][j]);
    }

    // Epilogue. C layout: VGPR e, lanes 0-15 -> M=e, lanes 16-31 -> M=e+8; N=lane%16.
    const int lh = lane >> 4;
    const int ln = lane & 15;
#pragma unroll
    for (int i = 0; i < 2; ++i)
#pragma unroll
        for (int j = 0; j < 4; ++j) {
            const int gmb = m0 + wm * 32 + i * 16 + lh * 8;
            const int gn  = n0 + wn * 64 + j * 16 + ln;
#pragma unroll
            for (int e = 0; e < 8; ++e) {
                const float val = acc[i][j][e];
                const int gm = gmb + e;
                if (MODE == 1) {
                    outF[(size_t)gm * 1024 + gn] = val;
                } else {
                    const unsigned short bv = f2bf(val);
                    const int part = gn >> 10;       // 0:q 1:k 2:v
                    const int cc   = gn & 1023;
                    const int h    = cc >> 6;
                    const int d    = cc & 63;
                    const int b    = gm >> 11;
                    const int t    = gm & 2047;
                    const int bh   = b * 16 + h;
                    if (part == 0)
                        qb[((size_t)bh * 2048 + t) * 64 + d] = bv;      // [BH][T][hd]
                    else if (part == 1)
                        kbuf[((size_t)bh * 2048 + t) * 64 + d] = bv;    // [BH][T][hd]
                    else
                        vb[((size_t)bh * 64 + d) * 2048 + t] = bv;      // [BH][hd][T]
                }
            }
        }
}

// ---------------------------------------------------------------------------
// Flash attention: grid (B*H, T/128). 8 waves/block, wave handles 16 q rows.
// Double-buffered K [64 keys][64 hd] and V^T [64 hd][64 keys] tiles staged by
// async global->LDS copies; online softmax in fp32 registers; P re-laid
// C->A through a per-wave LDS slab.
// ---------------------------------------------------------------------------
__global__ __launch_bounds__(256) void flash_attn(
    const unsigned short* __restrict__ qb,
    const unsigned short* __restrict__ kb,
    const unsigned short* __restrict__ vtb,
    unsigned short* __restrict__ yb) {
    __shared__ unsigned short Ks[2][64 * 72];
    __shared__ unsigned short Vs[2][64 * 72];
    __shared__ unsigned short Ps[8 * 16 * 72];

    const int tid  = threadIdx.x;
    const int lane = tid & 31;
    const int wave = tid >> 5;
    const int bh   = blockIdx.x;             // 0..63
    const int b    = bh >> 4;
    const int h    = bh & 15;
    const int q0b  = blockIdx.y * 128;
    const int qw   = q0b + wave * 16;        // wave's first t (q row)

    const unsigned short* qp = qb  + (size_t)bh * 2048 * 64;
    const unsigned short* kp = kb  + (size_t)bh * 2048 * 64;
    const unsigned short* vp = vtb + (size_t)bh * 64 * 2048;

    // Q A-fragments (rows qw..qw+15, hd chunks 0..31 / 32..63) straight from global
    bf16x16 qa[2];
#pragma unroll
    for (int kc = 0; kc < 2; ++kc) qa[kc] = ldFrag(qp + kc * 32, qw, lane, 64);

    const float NEG = -__builtin_inff();
    float  m8[8], l8[8];
    floatx8 o[4];
#pragma unroll
    for (int e = 0; e < 8; ++e) { m8[e] = NEG; l8[e] = 0.0f; }
#pragma unroll
    for (int f = 0; f < 4; ++f) o[f] = (floatx8)0.0f;

    const int lh   = lane >> 4;
    const int ln   = lane & 15;
    const int qmax = qw + 15;
    const int nkb  = (q0b >> 6) + 2;         // key blocks needed by this workgroup
    unsigned short* Pw = Ps + wave * 16 * 72;

    // async staging addresses: thread -> (row 0..63, col 0,16,32,48)
    const int srow = tid >> 2;
    const int scol = (tid & 3) * 16;
    const unsigned ksB = ldsOff(&Ks[0][0]) + (unsigned)(srow * 72 + scol) * 2u;
    const unsigned vsB = ldsOff(&Vs[0][0]) + (unsigned)(srow * 72 + scol) * 2u;
    const unsigned kvBytes = (unsigned)(64 * 72) * 2u;
    const unsigned short* gvRow = vp + (size_t)srow * 2048 + scol;

    // prologue: stage key-block 0 into buffer 0
    {
        const unsigned short* gk = kp + (size_t)srow * 64 + scol;
        asyncLoadB128(ksB,      gk);
        asyncLoadB128(ksB + 16, gk + 8);
        asyncLoadB128(vsB,      gvRow);
        asyncLoadB128(vsB + 16, gvRow + 8);
    }

    for (int kb0 = 0; kb0 < nkb; ++kb0) {
        const int kbase = kb0 * 64;
        waitAsync();
        __syncthreads();
        if (kb0 + 1 < nkb) {
            const unsigned nb = (unsigned)((kb0 + 1) & 1) * kvBytes;
            const unsigned short* gk =
                kp + (size_t)(kbase + 64 + srow) * 64 + scol;
            asyncLoadB128(ksB + nb,      gk);
            asyncLoadB128(ksB + nb + 16, gk + 8);
            asyncLoadB128(vsB + nb,      gvRow + kbase + 64);
            asyncLoadB128(vsB + nb + 16, gvRow + kbase + 64 + 8);
        }
        const unsigned short* ks = Ks[kb0 & 1];
        const unsigned short* vs = Vs[kb0 & 1];

        if (kbase <= qmax) {
            // S = (Q K^T) * 1/sqrt(hd), causal-masked
            floatx8 st[4];
#pragma unroll
            for (int j = 0; j < 4; ++j) {
                floatx8 s = (floatx8)0.0f;
                bf16x16 b0 = ldFrag(ks,      j * 16, lane, 72);   // hd 0..31
                bf16x16 b1 = ldFrag(ks + 32, j * 16, lane, 72);   // hd 32..63
                s = wmma_bf16(qa[0], b0, s);
                s = wmma_bf16(qa[1], b1, s);
                st[j] = s;
            }
            float rmax[8];
#pragma unroll
            for (int e = 0; e < 8; ++e) rmax[e] = NEG;
#pragma unroll
            for (int j = 0; j < 4; ++j)
#pragma unroll
                for (int e = 0; e < 8; ++e) {
                    const int mrow = qw + e + lh * 8;
                    const int ncol = kbase + j * 16 + ln;
                    float sv = st[j][e] * 0.125f;
                    if (ncol > mrow) sv = NEG;
                    st[j][e] = sv;
                    rmax[e] = fmaxf(rmax[e], sv);
                }
            // row-max across the 16 lanes holding one row (masks stay in-half)
#pragma unroll
            for (int e = 0; e < 8; ++e) {
                rmax[e] = fmaxf(rmax[e], __shfl_xor(rmax[e], 1, 32));
                rmax[e] = fmaxf(rmax[e], __shfl_xor(rmax[e], 2, 32));
                rmax[e] = fmaxf(rmax[e], __shfl_xor(rmax[e], 4, 32));
                rmax[e] = fmaxf(rmax[e], __shfl_xor(rmax[e], 8, 32));
            }
            float al[8], rs[8];
#pragma unroll
            for (int e = 0; e < 8; ++e) {
                const float nm = fmaxf(m8[e], rmax[e]);   // finite: block intersects row
                al[e] = __expf(m8[e] - nm);               // exp(-inf)=0 first time
                m8[e] = nm;
                rs[e] = 0.0f;
            }
#pragma unroll
            for (int f = 0; f < 4; ++f)
#pragma unroll
                for (int e = 0; e < 8; ++e) o[f][e] *= al[e];
            // P = exp(S - m), stash bf16 into per-wave LDS (C layout -> memory)
#pragma unroll
            for (int j = 0; j < 4; ++j)
#pragma unroll
                for (int e = 0; e < 8; ++e) {
                    const float p = __expf(st[j][e] - m8[e]);  // exp(-inf)=0 if masked
                    rs[e] += p;
                    Pw[(e + lh * 8) * 72 + j * 16 + ln] = f2bf(p);
                }
#pragma unroll
            for (int e = 0; e < 8; ++e) {
                rs[e] += __shfl_xor(rs[e], 1, 32);
                rs[e] += __shfl_xor(rs[e], 2, 32);
                rs[e] += __shfl_xor(rs[e], 4, 32);
                rs[e] += __shfl_xor(rs[e], 8, 32);
                l8[e] = l8[e] * al[e] + rs[e];
            }
            // O += P @ V  (A-fragments of P from LDS, B-fragments from V^T)
            bf16x16 pa0 = ldFrag(Pw,      0, lane, 72);   // keys 0..31
            bf16x16 pa1 = ldFrag(Pw + 32, 0, lane, 72);   // keys 32..63
#pragma unroll
            for (int f = 0; f < 4; ++f) {
                bf16x16 bv0 = ldFrag(vs,      f * 16, lane, 72);
                bf16x16 bv1 = ldFrag(vs + 32, f * 16, lane, 72);
                o[f] = wmma_bf16(pa0, bv0, o[f]);
                o[f] = wmma_bf16(pa1, bv1, o[f]);
            }
        }
    }

    // y[b, t, h*64 + d] = O / l   (bf16 for the projection GEMM)
    float inv[8];
#pragma unroll
    for (int e = 0; e < 8; ++e) inv[e] = (l8[e] > 0.0f) ? 1.0f / l8[e] : 0.0f;
#pragma unroll
    for (int f = 0; f < 4; ++f)
#pragma unroll
        for (int e = 0; e < 8; ++e) {
            const int t = qw + e + lh * 8;
            yb[((size_t)b * 2048 + t) * 1024 + h * 64 + f * 16 + ln] =
                f2bf(o[f][e] * inv[e]);
        }
}

// ---------------------------------------------------------------------------
// Host side
// ---------------------------------------------------------------------------
extern "C" void kernel_launch(void* const* d_in, const int* in_sizes, int n_in,
                              void* d_out, int out_size, void* d_ws, size_t ws_size,
                              hipStream_t stream) {
    const float* x  = (const float*)d_in[0];   // [4,2048,1024]
    const float* Wa = (const float*)d_in[1];   // [1024,3072]
    const float* Wp = (const float*)d_in[2];   // [1024,1024]
    float* out = (float*)d_out;                // [4,2048,1024]

    // bf16 scratch carve-out (shorts); total ~92.3 MB
    unsigned short* ws = (unsigned short*)d_ws;
    size_t off = 0;
    unsigned short* xb   = ws + off; off += (size_t)8192 * 1024;  // x bf16
    unsigned short* wat  = ws + off; off += (size_t)3072 * 1024;  // W_attn^T bf16
    unsigned short* wpt  = ws + off; off += (size_t)1024 * 1024;  // W_proj^T bf16
    unsigned short* qb   = ws + off; off += (size_t)64 * 2048 * 64;
    unsigned short* kbuf = ws + off; off += (size_t)64 * 2048 * 64;
    unsigned short* vb   = ws + off; off += (size_t)64 * 64 * 2048;
    unsigned short* yb   = ws + off; off += (size_t)8192 * 1024;
    (void)ws_size; (void)in_sizes; (void)n_in; (void)out_size;

    cvt_f32_bf16<<<(8192 * 1024) / (256 * 4), 256, 0, stream>>>(x, xb, 8192 * 1024);
    transpose_cvt<<<(3072 * 1024) / 256, 256, 0, stream>>>(Wa, wat, 1024, 3072);
    transpose_cvt<<<(1024 * 1024) / 256, 256, 0, stream>>>(Wp, wpt, 1024, 1024);

    gemm_bf16<0><<<dim3(64, 24), 256, 0, stream>>>(xb, wat, nullptr, qb, kbuf, vb, 1024);
    flash_attn<<<dim3(64, 16), 256, 0, stream>>>(qb, kbuf, vb, yb);
    gemm_bf16<1><<<dim3(64, 8), 256, 0, stream>>>(yb, wpt, out,
                                                  nullptr, nullptr, nullptr, 1024);
}
